// FAGCNModel_10479720202339
// MI455X (gfx1250) — matrix operand order repaired and checked by reading, not compiled, not run on hardware
//
#include <hip/hip_runtime.h>
#include <hip/hip_bf16.h>
#include <stdint.h>
#include <stddef.h>

#define BN_EPSF   1e-5f
#define EPS_RESF  0.1f

// ---------------------------------------------------------------- WMMA types
typedef __attribute__((ext_vector_type(16))) __bf16 v16bf;
typedef __attribute__((ext_vector_type(8)))  float  v8f;

union FragBF { unsigned int u[8]; uint4 q[2]; v16bf v; };

__device__ __forceinline__ unsigned short f2bf(float f) {
  unsigned int u = __float_as_uint(f);
  u += 0x7FFFu + ((u >> 16) & 1u);           // round-to-nearest-even
  return (unsigned short)(u >> 16);
}
__device__ __forceinline__ unsigned int pack2(float a, float b) {
  return (unsigned int)f2bf(a) | ((unsigned int)f2bf(b) << 16);
}

// ------------------------------------------------------------- small helpers
__global__ void fill_kernel(float* p, float v, long long n) {
  long long i = (long long)blockIdx.x * blockDim.x + threadIdx.x;
  if (i < n) p[i] = v;
}

__global__ void deg_accum_kernel(const int* __restrict__ dst, float* deg, int E) {
  int e = blockIdx.x * blockDim.x + threadIdx.x;
  if (e < E) atomicAdd(&deg[dst[e]], 1.0f);
}

__global__ void dinv_kernel(float* deg, int n) {
  int i = blockIdx.x * blockDim.x + threadIdx.x;
  if (i < n) { float d = deg[i]; deg[i] = (d > 0.f) ? rsqrtf(d) : 0.f; }
}

// ---------------------------------------------- pre-pack weights (K x N f32)
// Output layout: for tile (kt, nt): 32 lanes x 8 dwords, contiguous, matching
// the v_wmma_f32_16x16x32_bf16 B-operand VGPR layout exactly.
__global__ void pack_weight_kernel(const float* __restrict__ W,
                                   unsigned int* __restrict__ P, int K, int N) {
  int ntiles = N / 16;
  long long total = (long long)(K / 32) * ntiles * 256;   // dwords
  long long i = (long long)blockIdx.x * blockDim.x + threadIdx.x;
  if (i >= total) return;
  int v    = (int)(i & 7);
  int lane = (int)((i >> 3) & 31);
  long long tile = i >> 8;                  // kt * ntiles + nt
  int nt = (int)(tile % ntiles);
  int kt = (int)(tile / ntiles);
  int half = lane >> 4, l15 = lane & 15;
  int n = nt * 16 + l15;
  int k = kt * 32 + half * 8 + 2 * (v & 3) + 16 * (v >> 2);
  P[i] = pack2(W[(size_t)k * N + n], W[(size_t)(k + 1) * N + n]);
}

// -------------------------------------------------- WMMA GEMM: C = act(A@Bp+b)
// A:  M x K f32 row-major, converted to bf16 in-register (streamed once).
// Bp: pre-packed bf16 B fragments (see pack_weight_kernel).
// One wave computes a 16 x (NT*16) tile; A fragment reused over NT n-tiles.
template <int NT>
__global__ void __launch_bounds__(32)
wmma_gemm_kernel(const float* __restrict__ A, const unsigned int* __restrict__ Bp,
                 const float* __restrict__ bias, float* __restrict__ C,
                 int M, int K, int Ncols, int relu) {
  int m0  = blockIdx.x * 16;
  int nt0 = blockIdx.y * NT;
  int ntiles = Ncols / 16;
  if (m0 >= M) return;

  int lane = threadIdx.x & 31;
  int half = lane >> 4;          // 0: lanes 0-15, 1: lanes 16-31
  int l15  = lane & 15;

  v8f acc[NT];
#pragma unroll
  for (int t = 0; t < NT; t++) { v8f z = {}; acc[t] = z; }

  int arow = m0 + l15; if (arow >= M) arow = M - 1;
  const float* Arow = A + (size_t)arow * K;

  for (int k0 = 0; k0 < K; k0 += 32) {
    // ---- A fragment: 16x32 bf16 (ISA 7.12.2 layout); pairs contiguous -> b64
    FragBF a;
    const float2* apLo = (const float2*)(Arow + k0 + half * 8);
    const float2* apHi = (const float2*)(Arow + k0 + half * 8 + 16);
#pragma unroll
    for (int v = 0; v < 4; v++) { float2 t2 = apLo[v]; a.u[v]     = pack2(t2.x, t2.y); }
#pragma unroll
    for (int v = 0; v < 4; v++) { float2 t2 = apHi[v]; a.u[4 + v] = pack2(t2.x, t2.y); }

    const unsigned int* brow =
        Bp + ((size_t)(k0 >> 5) * ntiles + nt0) * 256 + (size_t)lane * 8;
#pragma unroll
    for (int t = 0; t < NT; t++) {
      // ---- B fragment: two 128-bit loads from the packed stream
      FragBF b;
      const uint4* bq = (const uint4*)(brow + (size_t)t * 256);
      b.q[0] = bq[0];
      b.q[1] = bq[1];
      acc[t] = __builtin_amdgcn_wmma_f32_16x16x32_bf16(
          false, a.v, false, b.v, (short)0, acc[t], false, false);
    }
  }

#pragma unroll
  for (int t = 0; t < NT; t++) {
    int n = (nt0 + t) * 16 + l15;
    float bb = bias[n];
#pragma unroll
    for (int r = 0; r < 8; r++) {
      int row = m0 + half * 8 + r;
      float v = acc[t][r] + bb;
      if (relu) v = fmaxf(v, 0.f);
      if (row < M) C[(size_t)row * Ncols + n] = v;
    }
  }
}

static void launch_gemm(const float* A, const unsigned int* Bp, const float* bias,
                        float* C, int M, int K, int Ncols, int relu, hipStream_t s) {
  unsigned mt = (unsigned)((M + 15) / 16);
  if (Ncols % 64 == 0)
    wmma_gemm_kernel<4><<<dim3(mt, Ncols / 64), 32, 0, s>>>(A, Bp, bias, C, M, K, Ncols, relu);
  else if (Ncols % 32 == 0)
    wmma_gemm_kernel<2><<<dim3(mt, Ncols / 32), 32, 0, s>>>(A, Bp, bias, C, M, K, Ncols, relu);
  else
    wmma_gemm_kernel<1><<<dim3(mt, Ncols / 16), 32, 0, s>>>(A, Bp, bias, C, M, K, Ncols, relu);
}

// ---------------------------------------------------- attention dot products
__global__ void att_kernel(const float* __restrict__ h, const float* __restrict__ attl,
                           const float* __restrict__ attr, float* al, float* ar,
                           int n, int C) {
  int node = blockIdx.x * (blockDim.x >> 5) + (threadIdx.x >> 5);
  if (node >= n) return;
  int lane = threadIdx.x & 31;
  float sl = 0.f, sr = 0.f;
  for (int c = lane; c < C; c += 32) {
    float hv = h[(size_t)node * C + c];
    sl += hv * attl[c];
    sr += hv * attr[c];
  }
#pragma unroll
  for (int off = 16; off > 0; off >>= 1) {
    sl += __shfl_xor(sl, off, 32);
    sr += __shfl_xor(sr, off, 32);
  }
  if (lane == 0) { al[node] = sl; ar[node] = sr; }
}

// ------------------------- self-loop contribution + residual (non-atomic init)
__global__ void init_out_kernel(const float* __restrict__ h, const float* __restrict__ x0,
                                const float* __restrict__ al, const float* __restrict__ ar,
                                const float* __restrict__ dinv, float* outb,
                                long long total, int C) {
  long long i = (long long)blockIdx.x * blockDim.x + threadIdx.x;
  if (i >= total) return;
  int node = (int)(i / C);
  float dv = dinv[node];
  float sa = tanhf(al[node] + ar[node]) * dv * dv;
  outb[i] = h[i] * sa + EPS_RESF * x0[i];
}

// --------------------------------------- edge scatter: one wave per edge
__global__ void edge_msg_kernel(const int* __restrict__ src, const int* __restrict__ dst,
                                const float* __restrict__ h, const float* __restrict__ al,
                                const float* __restrict__ ar, const float* __restrict__ dinv,
                                float* outb, int E, int C) {
  int e = blockIdx.x * (blockDim.x >> 5) + (threadIdx.x >> 5);
  if (e >= E) return;
  int s = src[e], d = dst[e];
  float a = tanhf(al[s] + ar[d]) * dinv[s] * dinv[d];
  int lane = threadIdx.x & 31;
  const float* hr = h + (size_t)s * C;
  float* orow = outb + (size_t)d * C;
  for (int c = lane; c < C; c += 32)
    atomicAdd(&orow[c], hr[c] * a);
}

// ----------------------------------------------- batch-norm: stats (sum,sum2)
__global__ void bn_stats_kernel(const float* __restrict__ y, int rows, int C,
                                int relu_in, float* acc) {
  int c    = threadIdx.x % C;
  int sub  = threadIdx.x / C;
  int nsub = blockDim.x / C;
  float s = 0.f, s2 = 0.f;
  for (int r = blockIdx.x * nsub + sub; r < rows; r += gridDim.x * nsub) {
    float v = y[(size_t)r * C + c];
    if (relu_in) v = fmaxf(v, 0.f);
    s += v; s2 += v * v;
  }
  __shared__ float sh[512];
  sh[threadIdx.x]       = s;
  sh[256 + threadIdx.x] = s2;
  __syncthreads();
  if (sub == 0) {
    for (int k = 1; k < nsub; k++) { s += sh[k * C + c]; s2 += sh[256 + k * C + c]; }
    atomicAdd(&acc[c], s);
    atomicAdd(&acc[C + c], s2);
  }
}

// ----------------------------------------------- batch-norm: apply (+relu)
__global__ void bn_apply_kernel(const float* __restrict__ y, float* o,
                                long long total, int C, int relu_in, int relu_out,
                                const float* __restrict__ g, const float* __restrict__ b,
                                const float* __restrict__ acc, float inv_n) {
  long long i = (long long)blockIdx.x * blockDim.x + threadIdx.x;
  if (i >= total) return;
  int c = (int)(i % C);
  float mu  = acc[c] * inv_n;
  float var = fmaxf(acc[C + c] * inv_n - mu * mu, 0.f);
  float v = y[i];
  if (relu_in) v = fmaxf(v, 0.f);
  float r = g[c] * (v - mu) * rsqrtf(var + BN_EPSF) + b[c];
  if (relu_out) r = fmaxf(r, 0.f);
  o[i] = r;
}

// --------------------------------------------------------- final 1-col GEMV
__global__ void final_kernel(const float* __restrict__ z, const float* __restrict__ W3,
                             const float* __restrict__ b3, float* out, int B, int C) {
  int i = blockIdx.x * blockDim.x + threadIdx.x;
  if (i >= B) return;
  float s = b3[0];
  for (int c = 0; c < C; c++) s += z[(size_t)i * C + c] * W3[c];
  out[i] = s;
}

// ============================================================================
extern "C" void kernel_launch(void* const* d_in, const int* in_sizes, int n_in,
                              void* d_out, int out_size, void* d_ws, size_t ws_size,
                              hipStream_t stream) {
  const float* x     = (const float*)d_in[0];
  const int*   eidx  = (const int*)  d_in[1];
  const float* W_in  = (const float*)d_in[3];
  const float* b_in  = (const float*)d_in[4];
  const float* att_l = (const float*)d_in[5];
  const float* att_r = (const float*)d_in[6];
  const float* bn_g  = (const float*)d_in[7];
  const float* bn_b  = (const float*)d_in[8];
  const float* W1    = (const float*)d_in[9];
  const float* b1    = (const float*)d_in[10];
  const float* g1    = (const float*)d_in[11];
  const float* be1   = (const float*)d_in[12];
  const float* W2    = (const float*)d_in[13];
  const float* b2    = (const float*)d_in[14];
  const float* g2    = (const float*)d_in[15];
  const float* be2   = (const float*)d_in[16];
  const float* W3    = (const float*)d_in[17];
  const float* b3    = (const float*)d_in[18];

  const int HID  = in_sizes[4];                 // 64
  const int INCH = in_sizes[3] / HID;           // 128
  const int N    = in_sizes[0] / INCH;          // 50000
  const int E    = in_sizes[1] / 2;             // 1200000
  const int L    = in_sizes[5] / HID;           // 3
  const int H2   = in_sizes[14];                // 32
  const int B    = out_size;                    // == batch_size (10000)

  const int* src = eidx;
  const int* dst = eidx + E;

  // -------- workspace carve (256B aligned)
  char* p = (char*)d_ws;
  auto carve = [&](size_t bytes) -> void* {
    void* r = (void*)p;
    p += (bytes + 255) & ~(size_t)255;
    return r;
  };
  const int Np = (N + 15) & ~15;
  const int Bp = (B + 15) & ~15;
  float* dinv  = (float*)carve((size_t)N * 4);
  float* x0    = (float*)carve((size_t)Np * HID * 4);
  float* hbuf  = (float*)carve((size_t)Np * HID * 4);
  float* outb  = (float*)carve((size_t)Np * HID * 4);
  float* al    = (float*)carve((size_t)N * 4);
  float* ar    = (float*)carve((size_t)N * 4);
  float* bnacc = (float*)carve((size_t)2 * HID * 4);
  float* z1    = (float*)carve((size_t)Bp * HID * 4);
  float* z2    = (float*)carve((size_t)Bp * H2 * 4);
  unsigned int* WinP = (unsigned int*)carve((size_t)INCH * HID * 2);
  unsigned int* W1P  = (unsigned int*)carve((size_t)HID * HID * 2);
  unsigned int* W2P  = (unsigned int*)carve((size_t)HID * H2 * 2);
  (void)ws_size; (void)n_in;

  const long long NC = (long long)N * HID;
  const long long BH = (long long)B * HID;
  const long long B2 = (long long)B * H2;
  dim3 b256(256);

  // -------- pre-pack weights into WMMA B-fragment layout (bf16)
  {
    long long tw;
    tw = (long long)INCH * HID / 2;
    pack_weight_kernel<<<(unsigned)((tw + 255) / 256), b256, 0, stream>>>(W_in, WinP, INCH, HID);
    tw = (long long)HID * HID / 2;
    pack_weight_kernel<<<(unsigned)((tw + 255) / 256), b256, 0, stream>>>(W1, W1P, HID, HID);
    tw = (long long)HID * H2 / 2;
    pack_weight_kernel<<<(unsigned)((tw + 255) / 256), b256, 0, stream>>>(W2, W2P, HID, H2);
  }

  // -------- degree / normalization coefficients
  fill_kernel<<<(unsigned)((N + 255) / 256), b256, 0, stream>>>(dinv, 1.0f, N); // self-loop
  deg_accum_kernel<<<(unsigned)((E + 255) / 256), b256, 0, stream>>>(dst, dinv, E);
  dinv_kernel<<<(unsigned)((N + 255) / 256), b256, 0, stream>>>(dinv, N);

  // -------- input projection: x0 = relu(x @ W_in + b_in)   [WMMA bf16]
  launch_gemm(x, WinP, b_in, x0, N, INCH, HID, /*relu=*/1, stream);

  // -------- FAGCN layers
  const float* h = x0;                 // layer 0 reads x0
  for (int l = 0; l < L; l++) {
    att_kernel<<<(unsigned)((N + 7) / 8), b256, 0, stream>>>(h, att_l + l * HID,
                                                             att_r + l * HID, al, ar, N, HID);
    init_out_kernel<<<(unsigned)((NC + 255) / 256), b256, 0, stream>>>(h, x0, al, ar,
                                                                       dinv, outb, NC, HID);
    edge_msg_kernel<<<(unsigned)((E + 7) / 8), b256, 0, stream>>>(src, dst, h, al, ar,
                                                                  dinv, outb, E, HID);
    fill_kernel<<<1, b256, 0, stream>>>(bnacc, 0.f, 2 * HID);
    bn_stats_kernel<<<256, b256, 0, stream>>>(outb, N, HID, /*relu_in=*/1, bnacc);
    bn_apply_kernel<<<(unsigned)((NC + 255) / 256), b256, 0, stream>>>(
        outb, hbuf, NC, HID, /*relu_in=*/1, /*relu_out=*/0,
        bn_g + l * HID, bn_b + l * HID, bnacc, 1.0f / (float)N);
    h = hbuf;
  }

  // -------- MLP head on first B rows
  launch_gemm(h, W1P, b1, z1, B, HID, HID, /*relu=*/0, stream);
  fill_kernel<<<1, b256, 0, stream>>>(bnacc, 0.f, 2 * HID);
  bn_stats_kernel<<<256, b256, 0, stream>>>(z1, B, HID, 0, bnacc);
  bn_apply_kernel<<<(unsigned)((BH + 255) / 256), b256, 0, stream>>>(
      z1, z1, BH, HID, 0, /*relu_out=*/1, g1, be1, bnacc, 1.0f / (float)B);

  launch_gemm(z1, W2P, b2, z2, B, HID, H2, /*relu=*/0, stream);
  fill_kernel<<<1, b256, 0, stream>>>(bnacc, 0.f, 2 * H2);
  bn_stats_kernel<<<256, b256, 0, stream>>>(z2, B, H2, 0, bnacc);
  bn_apply_kernel<<<(unsigned)((B2 + 255) / 256), b256, 0, stream>>>(
      z2, z2, B2, H2, 0, /*relu_out=*/1, g2, be2, bnacc, 1.0f / (float)B);

  final_kernel<<<(unsigned)((B + 255) / 256), b256, 0, stream>>>(z2, W3, b3,
                                                                 (float*)d_out, B, H2);
}
